// LSTM_Regression_5901285065273
// MI455X (gfx1250) — compile-verified, matrix operand order checked by reading, not run on hardware
//
#include <hip/hip_runtime.h>

// 2-layer LSTM (H=30) + linear head, batch 512, T=2048, fp16 WMMA formulation.
//
// Transposed recurrence: pre^T = Wpad @ h^T, D = (gate x batch) 16x16 tiles.
// - 4 gates padded to 32 rows each -> 128 padded gate rows = 8 M-tiles.
// - K (hidden) padded 30 -> 32; K-rows 30/31 of h1 carry {x_t, 1.0} and the
//   matching A-columns carry {w_ih1, b_ih1+b_hh1}, fusing input + bias into
//   the same WMMA chain. Layer-2 bias rides h2's K-row 31 the same way.
// - WMMA D layout (lane = batch col, vgpr = gate row) re-packs into the next
//   step's B operand (h^T) purely per-lane: B.half[p] = cvt_f16(d[p/8][p%8]).
//   => no transposes anywhere in the 2048-step loop.
// - Activations use the gfx1250 hardware v_tanh_f32 (sigmoid = 0.5+0.5*tanh(x/2));
//   fallback is exp + v_rcp_f32. Never the precise-division expansion.

#define HSZ   30
#define MPAD  128   // 4 gates x 32 padded rows
#define KPAD  32
#define TLEN  2048
#define BSZ   512

typedef __attribute__((ext_vector_type(16))) _Float16 v16h;
typedef __attribute__((ext_vector_type(8)))  _Float16 v8h;
typedef __attribute__((ext_vector_type(8)))  float    v8f;

#if __has_builtin(__builtin_amdgcn_tanhf)
#define HW_TANH __builtin_amdgcn_tanhf
#elif __has_builtin(__builtin_amdgcn_tanh_f32)
#define HW_TANH __builtin_amdgcn_tanh_f32
#endif

#ifdef HW_TANH
__device__ __forceinline__ float tanh_act(float x) { return HW_TANH(x); }
__device__ __forceinline__ float sig_act(float x)  { return fmaf(0.5f, HW_TANH(0.5f * x), 0.5f); }
#else
// exp + single v_rcp_f32 (no IEEE division fixup chain)
__device__ __forceinline__ float tanh_act(float x) {
  return fmaf(2.0f, __builtin_amdgcn_rcpf(1.0f + __expf(-2.0f * x)), -1.0f);
}
__device__ __forceinline__ float sig_act(float x) {
  return __builtin_amdgcn_rcpf(1.0f + __expf(-x));
}
#endif

// Load one 16x32 f16 A-tile from LDS into WMMA A layout.
// Lane L holds row m = L%16; halves 0..7 = k in [8*(L/16), +8),
// halves 8..15 = k in [16 + 8*(L/16), +8). Row-major LDS => two b128 loads.
__device__ __forceinline__ v16h load_a_tile(const _Float16* __restrict__ base,
                                            int tile, int lane) {
  const int m  = lane & 15;
  const int kb = (lane >> 4) << 3;
  const _Float16* row = base + ((tile << 4) + m) * KPAD;
  v8h lo = *(const v8h*)(row + kb);
  v8h hi = *(const v8h*)(row + 16 + kb);
  v16h a;
#pragma unroll
  for (int i = 0; i < 8; ++i) { a[i] = lo[i]; a[8 + i] = hi[i]; }
  return a;
}

__global__ __launch_bounds__(32, 1)
void lstm2_wmma_kernel(const float* __restrict__ x,
                       const float* __restrict__ w_ih1, const float* __restrict__ w_hh1,
                       const float* __restrict__ b_ih1, const float* __restrict__ b_hh1,
                       const float* __restrict__ w_ih2, const float* __restrict__ w_hh2,
                       const float* __restrict__ b_ih2, const float* __restrict__ b_hh2,
                       const float* __restrict__ w_fc,  const float* __restrict__ b_fc,
                       float* __restrict__ out) {
  __shared__ _Float16 sA1[MPAD * KPAD];   // layer1: W_hh1 | col30=w_ih1 | col31=b1
  __shared__ _Float16 sAi2[MPAD * KPAD];  // layer2 input-side: W_ih2 | cols30,31=0
  __shared__ _Float16 sAh2[MPAD * KPAD];  // layer2 recur-side: W_hh2 | col31=b2

  const int lane = threadIdx.x;           // one wave32 per workgroup
  const bool hi_half = lane >= 16;
  const int n = lane & 15;                // batch column within tile

  // ---- one-time: stage pre-swizzled padded weight matrices into LDS ----
  for (int idx = lane; idx < MPAD * KPAD; idx += 32) {
    const int m = idx >> 5, k = idx & 31;     // padded gate row, padded K col
    const int g = m >> 5, j = m & 31;         // gate id, element within gate
    const int row = g * HSZ + j;              // row in the 4H-major weights
    float a1 = 0.f, ai2 = 0.f, ah2 = 0.f;
    if (j < HSZ) {
      if (k < HSZ) {
        a1  = w_hh1[row * HSZ + k];
        ai2 = w_ih2[row * HSZ + k];
        ah2 = w_hh2[row * HSZ + k];
      } else if (k == 30) {
        a1  = w_ih1[row];                     // IN == 1
      } else {                                // k == 31: fused bias slot
        a1  = b_ih1[row] + b_hh1[row];
        ah2 = b_ih2[row] + b_hh2[row];
      }
    }
    sA1[idx]  = (_Float16)a1;
    sAi2[idx] = (_Float16)ai2;
    sAh2[idx] = (_Float16)ah2;
  }
  __syncthreads();

  // ---- FC weights in D layout (per-lane-half) ----
  float wfcD[2][8];
#pragma unroll
  for (int s = 0; s < 2; ++s)
#pragma unroll
    for (int r = 0; r < 8; ++r) {
      const int j = 16 * s + (hi_half ? 8 : 0) + r;
      wfcD[s][r] = (j < HSZ) ? w_fc[j] : 0.f;
    }
  const float bfc = b_fc[0];

  v8f zc;
#pragma unroll
  for (int i = 0; i < 8; ++i) zc[i] = 0.f;

  // ---- state ----
  v16h h1B, h2B;
#pragma unroll
  for (int i = 0; i < 16; ++i) { h1B[i] = (_Float16)0.f; h2B[i] = (_Float16)0.f; }
  if (hi_half) h2B[15] = (_Float16)1.f;     // layer2 bias rail (K=31)

  float c1[2][8], c2[2][8];
#pragma unroll
  for (int s = 0; s < 2; ++s)
#pragma unroll
    for (int r = 0; r < 8; ++r) { c1[s][r] = 0.f; c2[s][r] = 0.f; }

  const int bbase = blockIdx.x << 4;
  const float* xrow = x + (size_t)(bbase + n) * TLEN;   // IN == 1
  float* orow = out + (size_t)(bbase + n) * TLEN;       // OUT == 1
  float xcur = xrow[0];

  for (int t = 0; t < TLEN; ++t) {
    // software-pipelined input fetch for t+1 (latency hides under this step)
    const float xnext = xrow[min(t + 1, TLEN - 1)];

    // inject x_t and bias rail into layer-1 B operand (K rows 30,31)
    if (hi_half) { h1B[14] = (_Float16)xcur; h1B[15] = (_Float16)1.f; }

    // ---------------- layer 1: pre1^T = A1 @ h1^T  (x, b fused) ------------
    v16h h1Bn;
#pragma unroll
    for (int s = 0; s < 2; ++s) {           // j-subtile: s=0 -> j<16, s=1 -> j>=16
      v8f d[4];
#pragma unroll
      for (int g = 0; g < 4; ++g) {         // i, f, g, o at M-tiles 2g+s
        v16h a = load_a_tile(sA1, 2 * g + s, lane);
        d[g] = __builtin_amdgcn_wmma_f32_16x16x32_f16(false, a, false, h1B,
                                                      (short)0, zc, false, false);
      }
#pragma unroll
      for (int r = 0; r < 8; ++r) {
        const float iv = sig_act(d[0][r]);
        const float fv = sig_act(d[1][r]);
        const float gv = tanh_act(d[2][r]);
        const float ov = sig_act(d[3][r]);
        const float cn = fmaf(fv, c1[s][r], iv * gv);
        c1[s][r] = cn;
        h1Bn[8 * s + r] = (_Float16)(ov * tanh_act(cn));   // per-lane D->B pack
      }
    }

    // ------- layer 2: pre2^T = Ai2 @ h1^T + Ah2 @ h2^T  (b2 fused) ---------
    v16h h2Bn;
    float acc = 0.f;
#pragma unroll
    for (int s = 0; s < 2; ++s) {
      v8f d[4];
#pragma unroll
      for (int g = 0; g < 4; ++g) {
        v16h ah = load_a_tile(sAh2, 2 * g + s, lane);
        v8f tmp = __builtin_amdgcn_wmma_f32_16x16x32_f16(false, ah, false, h2B,
                                                         (short)0, zc, false, false);
        v16h ai = load_a_tile(sAi2, 2 * g + s, lane);
        d[g] = __builtin_amdgcn_wmma_f32_16x16x32_f16(false, ai, false, h1Bn,
                                                      (short)0, tmp, false, false);
      }
#pragma unroll
      for (int r = 0; r < 8; ++r) {
        const float iv = sig_act(d[0][r]);
        const float fv = sig_act(d[1][r]);
        const float gv = tanh_act(d[2][r]);
        const float ov = sig_act(d[3][r]);
        const float cn = fmaf(fv, c2[s][r], iv * gv);
        c2[s][r] = cn;
        const float hn = ov * tanh_act(cn);
        h2Bn[8 * s + r] = (_Float16)hn;
        acc = fmaf(hn, wfcD[s][r], acc);    // FC head partial (this lane's j's)
      }
    }
    if (hi_half) { h2Bn[14] = (_Float16)0.f; h2Bn[15] = (_Float16)1.f; }

    // FC head: combine the two lane-halves, store one f32 per batch row
    acc += __shfl_xor(acc, 16, 32);
    if (lane < 16) orow[t] = acc + bfc;

    h1B = h1Bn;
    h2B = h2Bn;
    xcur = xnext;
  }
}

extern "C" void kernel_launch(void* const* d_in, const int* in_sizes, int n_in,
                              void* d_out, int out_size, void* d_ws, size_t ws_size,
                              hipStream_t stream) {
  (void)in_sizes; (void)n_in; (void)out_size; (void)d_ws; (void)ws_size;
  const float* x     = (const float*)d_in[0];
  const float* w_ih1 = (const float*)d_in[1];
  const float* w_hh1 = (const float*)d_in[2];
  const float* b_ih1 = (const float*)d_in[3];
  const float* b_hh1 = (const float*)d_in[4];
  const float* w_ih2 = (const float*)d_in[5];
  const float* w_hh2 = (const float*)d_in[6];
  const float* b_ih2 = (const float*)d_in[7];
  const float* b_hh2 = (const float*)d_in[8];
  const float* w_fc  = (const float*)d_in[9];
  const float* b_fc  = (const float*)d_in[10];
  float* out = (float*)d_out;

  dim3 grid(BSZ / 16);   // 32 independent 16-row batch tiles, one wave32 each
  dim3 block(32);
  lstm2_wmma_kernel<<<grid, block, 0, stream>>>(x, w_ih1, w_hh1, b_ih1, b_hh1,
                                                w_ih2, w_hh2, b_ih2, b_hh2,
                                                w_fc, b_fc, out);
}
